// GraphHead_72327249264834
// MI455X (gfx1250) — matrix operand order repaired and checked by reading, not compiled, not run on hardware
//
#include <hip/hip_runtime.h>
#include <hip/hip_bf16.h>

// ---------------------------------------------------------------------------
// GINE GNN on gfx1250: graph phase = L2-resident gather/atomic-scatter,
// dense phase = f16 WMMA (v_wmma_f32_16x16x32_f16) GEMMs with f32 accumulate.
// ---------------------------------------------------------------------------

typedef _Float16 v16h __attribute__((ext_vector_type(16)));
typedef _Float16 v8h  __attribute__((ext_vector_type(8)));
typedef float    v8f  __attribute__((ext_vector_type(8)));

#define DIM 128
#define LDS_STRIDE 136   // halves; 272B row pitch -> lane n hits bank group 4n

// ---------------- Feature encoder: x[i,:] = node_emb[node_type[i],:] -------
__global__ void encode_nodes(const int* __restrict__ ntype,
                             const float* __restrict__ nemb,
                             float* __restrict__ x, _Float16* __restrict__ xh,
                             int total) {
    int i = blockIdx.x * blockDim.x + threadIdx.x;
    if (i >= total) return;
    int node = i >> 7, c = i & 127;
    float v = nemb[ntype[node] * DIM + c];
    x[i] = v;
    xh[i] = (_Float16)v;
}

// ---------------- Message + scatter: agg[dst] += relu(x[src] + e_emb[et]) --
// 32 lanes per edge, float4 per lane; x/agg stay L2-resident (25.6MB each).
__global__ void scatter_edges(const float* __restrict__ x,
                              const int* __restrict__ src,
                              const int* __restrict__ dst,
                              const int* __restrict__ etype,
                              const float* __restrict__ eemb,
                              float* __restrict__ agg, int E) {
    int tid = blockIdx.x * blockDim.x + threadIdx.x;
    int e = tid >> 5;
    if (e >= E) return;
    int q = (tid & 31) * 4;
    int s = src[e], d = dst[e], t = etype[e];
    float4 xv = *(const float4*)(x + (size_t)s * DIM + q);
    float4 ev = *(const float4*)(eemb + (size_t)t * DIM + q);
    float4 m;
    m.x = fmaxf(xv.x + ev.x, 0.f);
    m.y = fmaxf(xv.y + ev.y, 0.f);
    m.z = fmaxf(xv.z + ev.z, 0.f);
    m.w = fmaxf(xv.w + ev.w, 0.f);
    float* ap = agg + (size_t)d * DIM + q;
    atomicAdd(ap + 0, m.x);
    atomicAdd(ap + 1, m.y);
    atomicAdd(ap + 2, m.z);
    atomicAdd(ap + 3, m.w);
}

// ---------------- t = (1+eps)*x + agg, to f16 (WMMA A operand) -------------
__global__ void prep_a(const float* __restrict__ x, const float* __restrict__ agg,
                       const float* __restrict__ eps, int l,
                       _Float16* __restrict__ th, int total) {
    int i = blockIdx.x * blockDim.x + threadIdx.x;
    if (i >= total) return;
    float e = 1.f + eps[l];
    th[i] = (_Float16)(e * x[i] + agg[i]);
}

// ---------------- WMMA GEMM: out = act(A[rows,128] @ W[128,128] + b) -------
// Block = 8 waves; each wave owns a 16-row group x all 128 cols (8 C tiles).
// W (f32, row-major) is converted to f16 transposed into LDS once per block.
template <bool RELU, bool OUT_H, bool OUT_F>
__global__ void __launch_bounds__(256)
gemm_wmma(const _Float16* __restrict__ A, const float* __restrict__ W,
          const float* __restrict__ bias, _Float16* __restrict__ outH,
          float* __restrict__ outF, int rows) {
    __shared__ _Float16 wt[DIM * LDS_STRIDE];
    const int t = threadIdx.x;
#pragma unroll
    for (int i = 0; i < 64; ++i) {      // 16384 f32 -> f16 transposed
        int idx = t + i * 256;
        int r = idx >> 7, c = idx & 127;
        wt[c * LDS_STRIDE + r] = (_Float16)W[idx];
    }
    __syncthreads();

    const int wave = t >> 5;
    const int lane = t & 31;
    const int rg = blockIdx.x * 8 + wave;           // 16-row group
    if (rg * 16 >= rows) return;                    // wave-uniform: EXEC stays all-1s

    const int m = lane & 15;
    const int hiHalf = (lane >= 16);
    const int koffA = hiHalf ? 8 : 0;               // A: K {0..7,16..23} / {8..15,24..31}
    const int koffB = hiHalf ? 8 : 0;               // B: K {0..7,16..23} / {8..15,24..31}

    v8f acc[8];
#pragma unroll
    for (int nt = 0; nt < 8; ++nt) acc[nt] = v8f{0, 0, 0, 0, 0, 0, 0, 0};

    const _Float16* arow = A + (size_t)(rg * 16 + m) * DIM;
#pragma unroll
    for (int ks = 0; ks < 4; ++ks) {                // K = 128 in steps of 32
        const int kb = ks * 32;
        v8h alo = *(const v8h*)(arow + kb + koffA);
        v8h ahi = *(const v8h*)(arow + kb + koffA + 16);
        v16h a;
#pragma unroll
        for (int i = 0; i < 8; ++i) { a[i] = alo[i]; a[i + 8] = ahi[i]; }
#pragma unroll
        for (int nt = 0; nt < 8; ++nt) {            // column tiles
            const _Float16* bp = &wt[(nt * 16 + m) * LDS_STRIDE + kb + koffB];
            v8h blo = *(const v8h*)bp;
            v8h bhi = *(const v8h*)(bp + 16);
            v16h b;
#pragma unroll
            for (int i = 0; i < 8; ++i) { b[i] = blo[i]; b[i + 8] = bhi[i]; }
            acc[nt] = __builtin_amdgcn_wmma_f32_16x16x32_f16(
                false, a, false, b, (short)0, acc[nt], false, false);
        }
    }

    // C/D layout: VGPR v -> row v (lanes 0-15) or v+8 (lanes 16-31), col = lane%16
    const int rbase = rg * 16 + (hiHalf ? 8 : 0);
#pragma unroll
    for (int nt = 0; nt < 8; ++nt) {
        const int col = nt * 16 + m;
        const float bv = bias[col];
#pragma unroll
        for (int v = 0; v < 8; ++v) {
            float val = acc[nt][v] + bv;
            if (RELU) val = fmaxf(val, 0.f);
            const size_t o = (size_t)(rbase + v) * DIM + col;
            if constexpr (OUT_H) outH[o] = (_Float16)val;
            if constexpr (OUT_F) outF[o] = val;
        }
    }
}

// ---------------- BatchNorm: per-column sum / sumsq ------------------------
__global__ void bn_stats(const float* __restrict__ h, float* __restrict__ stats, int n) {
    int c = threadIdx.x;                 // 128 threads = 128 columns
    int per = (n + gridDim.x - 1) / gridDim.x;
    int r0 = blockIdx.x * per;
    int r1 = min(n, r0 + per);
    float s = 0.f, s2 = 0.f;
    for (int r = r0; r < r1; ++r) {
        float v = h[(size_t)r * DIM + c];
        s += v;
        s2 += v * v;
    }
    atomicAdd(&stats[c], s);
    atomicAdd(&stats[DIM + c], s2);
}

__global__ void bn_apply(const float* __restrict__ h, const float* __restrict__ stats,
                         const float* __restrict__ gamma, const float* __restrict__ beta,
                         float* __restrict__ x, _Float16* __restrict__ xh,
                         int total, float invN) {
    int i = blockIdx.x * blockDim.x + threadIdx.x;
    if (i >= total) return;
    int c = i & 127;
    float mu = stats[c] * invN;
    float var = stats[DIM + c] * invN - mu * mu;   // biased, matches jnp.var
    float v = (h[i] - mu) * rsqrtf(var + 1e-5f) * gamma[c] + beta[c];
    v = fmaxf(v, 0.f);
    x[i] = v;
    xh[i] = (_Float16)v;
}

// ---------------- Head: pred = h @ hW3 + hb3 (one wave per row) ------------
__global__ void head_out(const _Float16* __restrict__ h, const float* __restrict__ w,
                         const float* __restrict__ b, float* __restrict__ out, int n) {
    int wave = blockIdx.x * (blockDim.x >> 5) + (threadIdx.x >> 5);
    int lane = threadIdx.x & 31;
    if (wave >= n) return;
    const _Float16* row = h + (size_t)wave * DIM;
    float s = 0.f;
#pragma unroll
    for (int i = 0; i < 4; ++i) s += (float)row[lane + i * 32] * w[lane + i * 32];
    for (int off = 16; off > 0; off >>= 1) s += __shfl_down(s, off, 32);
    if (lane == 0) out[wave] = s + b[0];
}

// ---------------------------------------------------------------------------
static inline int cdiv(int a, int b) { return (a + b - 1) / b; }

extern "C" void kernel_launch(void* const* d_in, const int* in_sizes, int n_in,
                              void* d_out, int out_size, void* d_ws, size_t ws_size,
                              hipStream_t stream) {
    const int* node_type = (const int*)d_in[0];
    const int* edge_type = (const int*)d_in[1];
    const int* edge_index = (const int*)d_in[2];
    const float* node_emb = (const float*)d_in[3];
    const float* edge_emb = (const float*)d_in[4];
    const float* eps = (const float*)d_in[5];
    const float* W1 = (const float*)d_in[6];
    const float* b1 = (const float*)d_in[7];
    const float* W2 = (const float*)d_in[8];
    const float* b2 = (const float*)d_in[9];
    const float* gamma = (const float*)d_in[10];
    const float* beta = (const float*)d_in[11];
    const float* hW1 = (const float*)d_in[12];
    const float* hb1 = (const float*)d_in[13];
    const float* hW2 = (const float*)d_in[14];
    const float* hb2 = (const float*)d_in[15];
    const float* hW3 = (const float*)d_in[16];
    const float* hb3 = (const float*)d_in[17];
    float* out = (float*)d_out;

    const int N = in_sizes[0];
    const int E = in_sizes[1];
    const int total = N * DIM;
    const int* src = edge_index;
    const int* dst = edge_index + E;

    // Workspace carve (all regions 16B aligned)
    char* ws = (char*)d_ws;
    float* x = (float*)ws;                               // N*128 f32
    float* agg = (float*)(ws + (size_t)total * 4);       // N*128 f32
    float* h2 = (float*)(ws + (size_t)total * 8);        // N*128 f32
    float* stats = (float*)(ws + (size_t)total * 12);    // 256 f32
    char* hbase = ws + (size_t)total * 12 + 1024;
    _Float16* xh = (_Float16*)hbase;                     // N*128 f16
    _Float16* th = (_Float16*)(hbase + (size_t)total * 2);
    _Float16* h1h = (_Float16*)(hbase + (size_t)total * 4);

    const int gElem = cdiv(total, 256);
    const int gGemm = cdiv(cdiv(N, 16), 8);

    encode_nodes<<<gElem, 256, 0, stream>>>(node_type, node_emb, x, xh, total);

    for (int l = 0; l < 3; ++l) {
        hipMemsetAsync(agg, 0, (size_t)total * 4, stream);
        hipMemsetAsync(stats, 0, 256 * 4, stream);
        scatter_edges<<<cdiv(E * 32, 256), 256, 0, stream>>>(
            x, src, dst, edge_type, edge_emb, agg, E);
        prep_a<<<gElem, 256, 0, stream>>>(x, agg, eps, l, th, total);
        gemm_wmma<true, true, false><<<gGemm, 256, 0, stream>>>(
            th, W1 + (size_t)l * DIM * DIM, b1 + l * DIM, h1h, (float*)nullptr, N);
        gemm_wmma<false, false, true><<<gGemm, 256, 0, stream>>>(
            h1h, W2 + (size_t)l * DIM * DIM, b2 + l * DIM, (_Float16*)nullptr, h2, N);
        bn_stats<<<128, 128, 0, stream>>>(h2, stats, N);
        bn_apply<<<gElem, 256, 0, stream>>>(h2, stats, gamma + l * DIM, beta + l * DIM,
                                            x, xh, total, 1.f / (float)N);
    }

    // Head MLP: 128 -> 128 -> 128 -> 1
    gemm_wmma<true, true, false><<<gGemm, 256, 0, stream>>>(
        xh, hW1, hb1, h1h, (float*)nullptr, N);
    gemm_wmma<true, true, false><<<gGemm, 256, 0, stream>>>(
        h1h, hW2, hb2, th, (float*)nullptr, N);
    head_out<<<cdiv(N, 8), 256, 0, stream>>>(th, hW3, hb3, out, N);
}
